// AtomCenteredTensorMomentDescriptor_16673063043102
// MI455X (gfx1250) — matrix operand order repaired and checked by reading, not compiled, not run on hardware
//
#include <hip/hip_runtime.h>
#include <math.h>

#define SH_DIM 25
#define NF 64
#define RCUT 5.0f
#define NRAD 64
#define CAP_K 640
#define ENT_CAP 3072       // max packed CG entries cached in LDS (~1.5K actual)
#define RSTR 72            // padded bf16 halves per staging row (conflict avoidance)
#define NW 2               // waves per block (64 threads, wave32)

typedef __attribute__((ext_vector_type(16))) __bf16 v16bf;
typedef __attribute__((ext_vector_type(8)))  float  v8f;

union V16 { v16bf v; uint4 q[2]; };

// Native f32->bf16 conversion (RTNE) -- lowers to hardware cvt on gfx1250,
// replacing the 4-op manual rounding emulation.
__device__ __forceinline__ unsigned short f2bf(float x) {
  union { __bf16 h; unsigned short s; } u;
  u.h = (__bf16)x;
  return u.s;
}
__device__ __forceinline__ float bf2f(unsigned short h) {
  return __uint_as_float(((unsigned)h) << 16);
}
__device__ __forceinline__ int deg_of(int i) {
  int l = 0;
  while ((l + 1) * (l + 1) <= i) ++l;
  return l;
}

__device__ __forceinline__ v8f wmma_bf16(const V16& a, const V16& b, v8f c) {
  return __builtin_amdgcn_wmma_f32_16x16x32_bf16(false, a.v, false, b.v,
                                                 (short)0, c, false, false);
}

// ---- A fragment (16x32 bf16) from LDS staging tile, per ISA 16-bit A layout ----
// lanes 0-15: M=lane, K {0..7,16..23}; lanes 16-31: M=lane-16, K {8..15,24..31}
__device__ __forceinline__ void load_afrag(const unsigned short* base, int lane,
                                           int kc, V16& f) {
  int m = lane & 15, g = lane >> 4;
  const unsigned short* p = base + m * RSTR + kc * 32 + g * 8;
  f.q[0] = *(const uint4*)p;          // K +0..7   -> V0..V3
  f.q[1] = *(const uint4*)(p + 16);   // K +16..23 -> V4..V7
}
// ---- B fragment (32x16 bf16) from N-major weight tile (WT[n][k]) ----
// lanes 0-15: N=lane, K=kc*32+{0..7,16..23}; lanes 16-31: N=lane-16, K=kc*32+{8..15,24..31}
__device__ __forceinline__ void load_bfrag(const unsigned short* W, int lane,
                                           int nt, int kc, V16& f) {
  int n = lane & 15, g = lane >> 4;
  const unsigned short* p = W + (nt * 16 + n) * 64 + kc * 32 + g * 8;
  f.q[0] = *(const uint4*)p;
  f.q[1] = *(const uint4*)(p + 16);
}

// ======================= device-side CG table construction =======================
__device__ double d_fact(int n) { double r = 1.0; for (int i = 2; i <= n; ++i) r *= (double)i; return r; }

__device__ double cg_complex(int l1, int m1, int l2, int m2, int l3, int m3) {
  int dl = (l1 > l2) ? (l1 - l2) : (l2 - l1);
  if (m1 + m2 != m3 || l3 < dl || l3 > l1 + l2) return 0.0;
  double pref = sqrt((2.0 * l3 + 1.0) * d_fact(l1 + l2 - l3) * d_fact(l1 - l2 + l3) *
                     d_fact(-l1 + l2 + l3) / d_fact(l1 + l2 + l3 + 1));
  pref *= sqrt(d_fact(l1 + m1) * d_fact(l1 - m1) * d_fact(l2 + m2) * d_fact(l2 - m2) *
               d_fact(l3 + m3) * d_fact(l3 - m3));
  double s = 0.0;
  for (int k = 0; k <= l1 + l2 - l3; ++k) {
    int d2 = l1 - m1 - k, d3 = l2 + m2 - k, d4 = l3 - l2 + m1 + k, d5 = l3 - l1 - m2 + k;
    if (d2 < 0 || d3 < 0 || d4 < 0 || d5 < 0) continue;
    double den = d_fact(k) * d_fact(l1 + l2 - l3 - k) * d_fact(d2) * d_fact(d3) *
                 d_fact(d4) * d_fact(d5);
    s += ((k & 1) ? -1.0 : 1.0) / den;
  }
  return pref * s;
}

struct C2 { double re, im; };
// Row r of the complex->real SH transform U_l has at most 2 nonzeros.
__device__ int u_row(int l, int r, C2* u, int* col) {
  const double rs2 = 0.70710678118654752440;
  if (r == l) { u[0].re = 1.0; u[0].im = 0.0; col[0] = l; return 1; }
  if (r > l) {
    int m = r - l; double s = (m & 1) ? -1.0 : 1.0;
    u[0].re = s * rs2; u[0].im = 0.0; col[0] = l + m;
    u[1].re = rs2;     u[1].im = 0.0; col[1] = l - m;
    return 2;
  }
  int m = l - r; double s = (m & 1) ? -1.0 : 1.0;
  u[0].re = 0.0; u[0].im = rs2;       col[0] = l - m;
  u[1].re = 0.0; u[1].im = -s * rs2;  col[1] = l + m;
  return 2;
}

__global__ void cg_dense_kernel(float* __restrict__ C) {
  int idx = blockIdx.x * blockDim.x + threadIdx.x;
  if (idx >= SH_DIM * SH_DIM * SH_DIM) return;
  int i = idx / (SH_DIM * SH_DIM);
  int j = (idx / SH_DIM) % SH_DIM;
  int k = idx % SH_DIM;
  int l1 = deg_of(i), l2 = deg_of(j), l3 = deg_of(k);
  int a = i - l1 * l1, b = j - l2 * l2, c = k - l3 * l3;
  C2 u1[2], u2[2], u3[2]; int c1[2], c2[2], c3[2];
  int n1 = u_row(l1, a, u1, c1), n2 = u_row(l2, b, u2, c2), n3 = u_row(l3, c, u3, c3);
  double re = 0.0, im = 0.0;
  for (int t1 = 0; t1 < n1; ++t1)
    for (int t2 = 0; t2 < n2; ++t2)
      for (int t3 = 0; t3 < n3; ++t3) {
        double cg = cg_complex(l1, c1[t1] - l1, l2, c2[t2] - l2, l3, c3[t3] - l3);
        if (cg == 0.0) continue;
        C2 p; // u1*u2
        p.re = u1[t1].re * u2[t2].re - u1[t1].im * u2[t2].im;
        p.im = u1[t1].re * u2[t2].im + u1[t1].im * u2[t2].re;
        C2 q; // p * conj(u3)
        q.re = p.re * u3[t3].re + p.im * u3[t3].im;
        q.im = p.im * u3[t3].re - p.re * u3[t3].im;
        re += q.re * cg; im += q.im * cg;
      }
  double val = ((l1 + l2 + l3) & 1) ? im : re;
  C[idx] = (float)val;
}

// Builds per-k strided lists, then a packed list + prefix offsets (single wave).
__global__ void cg_sparse_kernel(const float* __restrict__ C,
                                 uint2* __restrict__ strided,
                                 uint2* __restrict__ packed,
                                 int* __restrict__ koff) {
  __shared__ int scnt[SH_DIM];
  int k = threadIdx.x;
  if (k < SH_DIM) {
    int n = 0;
    for (int i = 0; i < SH_DIM; ++i)
      for (int j = 0; j < SH_DIM; ++j) {
        float v = C[(i * SH_DIM + j) * SH_DIM + k];
        if (fabsf(v) > 1e-6f) {
          unsigned pk = (unsigned)i | ((unsigned)j << 5) |
                        ((unsigned)deg_of(i) << 10) | ((unsigned)deg_of(j) << 13);
          strided[k * CAP_K + n] = make_uint2(pk, __float_as_uint(v));
          ++n;
        }
      }
    scnt[k] = n;
  }
  __syncthreads();
  if (k == 0) {
    int off = 0;
    koff[0] = 0;
    for (int q = 0; q < SH_DIM; ++q) {
      int nn = scnt[q];
      if (off + nn > ENT_CAP) nn = ENT_CAP - off;
      off += nn;
      koff[q + 1] = off;
    }
  }
  __syncthreads();
  if (k < SH_DIM) {
    int base = koff[k];
    int nn = koff[k + 1] - base;
    for (int t = 0; t < nn; ++t) packed[base + t] = strided[k * CAP_K + t];
  }
}

// ======================= weight transpose + bf16 conversion =======================
__global__ void prep_weights_kernel(const float* __restrict__ W1, const float* __restrict__ W2,
                                    const float* __restrict__ Wt,
                                    unsigned short* __restrict__ wt1T,
                                    unsigned short* __restrict__ wt2T,
                                    unsigned short* __restrict__ wtT) {
  int e = blockIdx.x * blockDim.x + threadIdx.x;
  if (e < 5 * 4096) {                        // W[l][f][g] -> WT[l][g][f]
    int l = e >> 12, rem = e & 4095, f = rem >> 6, g = rem & 63;
    wt1T[(l << 12) + (g << 6) + f] = f2bf(W1[e]);
    wt2T[(l << 12) + (g << 6) + f] = f2bf(W2[e]);
  } else {
    int e2 = e - 5 * 4096;
    if (e2 < 4096) { int f = e2 >> 6, g = e2 & 63; wtT[(g << 6) + f] = f2bf(Wt[e2]); }
  }
}

// ======================= real spherical harmonics (l<=4) =======================
__device__ void real_sh(float x, float y, float z, float* Y) {
  float x2 = x * x, y2 = y * y, z2 = z * z;
  Y[0] = 0.28209479177387814f;
  Y[1] = 0.4886025119029199f * y;
  Y[2] = 0.4886025119029199f * z;
  Y[3] = 0.4886025119029199f * x;
  Y[4] = 1.0925484305920792f * x * y;
  Y[5] = 1.0925484305920792f * y * z;
  Y[6] = 0.31539156525252005f * (3.0f * z2 - 1.0f);
  Y[7] = 1.0925484305920792f * x * z;
  Y[8] = 0.5462742152960396f * (x2 - y2);
  Y[9] = 0.5900435899266435f * y * (3.0f * x2 - y2);
  Y[10] = 2.890611442640554f * x * y * z;
  Y[11] = 0.4570457994644658f * y * (5.0f * z2 - 1.0f);
  Y[12] = 0.3731763325901154f * z * (5.0f * z2 - 3.0f);
  Y[13] = 0.4570457994644658f * x * (5.0f * z2 - 1.0f);
  Y[14] = 1.445305721320277f * z * (x2 - y2);
  Y[15] = 0.5900435899266435f * x * (x2 - 3.0f * y2);
  Y[16] = 2.5033429417967046f * x * y * (x2 - y2);
  Y[17] = 1.7701307697799304f * y * z * (3.0f * x2 - y2);
  Y[18] = 0.9461746957575601f * x * y * (7.0f * z2 - 1.0f);
  Y[19] = 0.6690465435572892f * y * z * (7.0f * z2 - 3.0f);
  Y[20] = 0.10578554691520431f * (35.0f * z2 * z2 - 30.0f * z2 + 3.0f);
  Y[21] = 0.6690465435572892f * x * z * (7.0f * z2 - 3.0f);
  Y[22] = 0.47308734787878004f * (x2 - y2) * (7.0f * z2 - 1.0f);
  Y[23] = 1.7701307697799304f * x * z * (x2 - 3.0f * y2);
  Y[24] = 0.6258357354491761f * (x2 * x2 - 6.0f * x2 * y2 + y2 * y2);
}

// ======================= main fused pair kernel =======================
__global__ __launch_bounds__(32 * NW, 1) void pair_kernel(
    const int* __restrict__ an, const float* __restrict__ disp,
    const int* __restrict__ nidx, const float* __restrict__ species,
    const float* __restrict__ bt, const float* __restrict__ bias1,
    const float* __restrict__ bias2,
    const unsigned short* __restrict__ wt1T, const unsigned short* __restrict__ wt2T,
    const unsigned short* __restrict__ wtT,
    const uint2* __restrict__ ent_packed, const int* __restrict__ koff_g,
    float* __restrict__ out, int num_pairs, int num_atoms) {
  // Per-wave-private staging regions + one block-shared CG entry cache.
  __shared__ unsigned short sIn[NW][16][RSTR];      // bf16 staging (emb rows, then radial)
  __shared__ float          sY[NW][16][26];         // spherical harmonics
  __shared__ unsigned short sA[NW][5][16][64];      // radial @ W1[l], bf16 (bias folded, l=0)
  __shared__ unsigned short sB[NW][5][16][64];      // radial @ W2[l], bf16 (bias folded, l=0)
  __shared__ unsigned short sEmb[NW][16][64];       // embedding gate, bf16
  __shared__ int            sKoff[26];
  __shared__ uint2          sEnt[ENT_CAP];          // packed CG table (broadcast reads)

  int w = threadIdx.x >> 5, lane = threadIdx.x & 31;

  // ---- cache the CG sparse table in LDS (block cooperative) ----
  if (threadIdx.x < 26) sKoff[threadIdx.x] = koff_g[threadIdx.x];
  __syncthreads();
  {
    int tot = sKoff[25];
    for (int t = threadIdx.x; t < tot; t += 32 * NW) sEnt[t] = ent_packed[t];
  }
  __syncthreads();

  int pbase = (blockIdx.x * NW + w) * 16;
  int p2 = lane >> 1, half = lane & 1;
  int pair = min(pbase + p2, num_pairs - 1);

  float dx = disp[3 * pair + 0], dy = disp[3 * pair + 1], dz = disp[3 * pair + 2];
  float r = sqrtf(dx * dx + dy * dy + dz * dz + 1e-12f);
  const float delta = RCUT / (float)(NRAD - 1);
  float env = 0.5f * (__cosf(3.14159265358979f * fminf(r / RCUT, 1.0f)) + 1.0f);
  int aj = an[nidx[num_pairs + pair]];
  int ai = an[nidx[pair]];
  const float* sej = species + aj * NF + half * 32;
  const float* sei = species + ai * NF + half * 32;

  // ---- phase 1a: stage gathered species rows (for the embedding gate GEMM) ----
  {
    unsigned wb[16];
#pragma unroll
    for (int c = 0; c < 16; ++c) {
      unsigned lo = f2bf(sei[2 * c]), hi = f2bf(sei[2 * c + 1]);
      wb[c] = lo | (hi << 16);
    }
    uint4* dst = (uint4*)&sIn[w][p2][half * 32];
    dst[0] = make_uint4(wb[0], wb[1], wb[2], wb[3]);
    dst[1] = make_uint4(wb[4], wb[5], wb[6], wb[7]);
    dst[2] = make_uint4(wb[8], wb[9], wb[10], wb[11]);
    dst[3] = make_uint4(wb[12], wb[13], wb[14], wb[15]);
  }
  // spherical harmonics (lanes 0..15, one pair each)
  if (lane < 16) {
    int pr = min(pbase + lane, num_pairs - 1);
    float ax = disp[3 * pr + 0], ay = disp[3 * pr + 1], az = disp[3 * pr + 2];
    float rr = sqrtf(ax * ax + ay * ay + az * az + 1e-12f);
    float Yv[SH_DIM];
    real_sh(ax / rr, ay / rr, az / rr, Yv);
#pragma unroll
    for (int q = 0; q < SH_DIM; ++q) sY[w][lane][q] = Yv[q];
  }

  // ---- phase 1b: embedding gate GEMM: emb = sE[idx_i] @ Wt + bt ----
  {
    V16 a0, a1;
    load_afrag(&sIn[w][0][0], lane, 0, a0);
    load_afrag(&sIn[w][0][0], lane, 1, a1);
#pragma unroll
    for (int nt = 0; nt < 4; ++nt) {
      v8f c = {};
      V16 bf0, bf1;
      load_bfrag(wtT, lane, nt, 0, bf0);
      load_bfrag(wtT, lane, nt, 1, bf1);
      c = wmma_bf16(a0, bf0, c);
      c = wmma_bf16(a1, bf1, c);
      int n = nt * 16 + (lane & 15);
      float btv = bt[n];
#pragma unroll
      for (int v = 0; v < 8; ++v) {
        int row = v + 8 * (lane >> 4);
        sEmb[w][row][n] = f2bf(c[v] + btv);
      }
    }
  }

  // ---- phase 2a: radial tile (reuses sIn; emb A-frags already consumed) ----
  {
    float inv2 = 0.5f / (delta * delta);
    unsigned wb[16];
#pragma unroll
    for (int c = 0; c < 16; ++c) {
      float r0 = r - (float)(half * 32 + 2 * c) * delta;
      float r1 = r - (float)(half * 32 + 2 * c + 1) * delta;
      float g0 = __expf(-inv2 * r0 * r0) * env * sej[2 * c];
      float g1 = __expf(-inv2 * r1 * r1) * env * sej[2 * c + 1];
      wb[c] = (unsigned)f2bf(g0) | ((unsigned)f2bf(g1) << 16);
    }
    uint4* dst = (uint4*)&sIn[w][p2][half * 32];
    dst[0] = make_uint4(wb[0], wb[1], wb[2], wb[3]);
    dst[1] = make_uint4(wb[4], wb[5], wb[6], wb[7]);
    dst[2] = make_uint4(wb[8], wb[9], wb[10], wb[11]);
    dst[3] = make_uint4(wb[12], wb[13], wb[14], wb[15]);
  }

  // ---- phase 2b: 10 degree GEMMs: A_l = radial @ W1[l], B_l = radial @ W2[l] ----
  // lm=0 bias is folded into the l=0 slab: A'_0 = A_0 + b/Y00, so the hot loop
  // later needs no bias handling (Y00 is the constant monopole harmonic).
  {
    V16 a0, a1;
    load_afrag(&sIn[w][0][0], lane, 0, a0);
    load_afrag(&sIn[w][0][0], lane, 1, a1);
    const float INV_Y00 = 3.5449077018110318f;   // 1 / 0.28209479177387814
#pragma unroll
    for (int mat = 0; mat < 2; ++mat) {
      const unsigned short* Wm = mat ? wt2T : wt1T;
      unsigned short* Sdst = mat ? &sB[w][0][0][0] : &sA[w][0][0][0];
      const float* bm = mat ? bias2 : bias1;
#pragma unroll
      for (int l = 0; l < 5; ++l) {
#pragma unroll
        for (int nt = 0; nt < 4; ++nt) {
          v8f c = {};
          V16 bf0, bf1;
          load_bfrag(Wm + l * 4096, lane, nt, 0, bf0);
          load_bfrag(Wm + l * 4096, lane, nt, 1, bf1);
          c = wmma_bf16(a0, bf0, c);
          c = wmma_bf16(a1, bf1, c);
          int n = nt * 16 + (lane & 15);
          float badd = (l == 0) ? bm[n] * INV_Y00 : 0.0f;
#pragma unroll
          for (int v = 0; v < 8; ++v) {
            int row = v + 8 * (lane >> 4);
            Sdst[(l * 16 + row) * 64 + n] = f2bf(c[v] + badd);
          }
        }
      }
    }
  }

  // ---- phase 3: sparse CG recombination + gate + atomic scatter ----
  // Both feature halves per entry traversal; per-entry uniform work amortized.
  for (int pp = 0; pp < 16; ++pp) {
    int pr = pbase + pp;
    if (pr >= num_pairs) break;
    float* obase = out + (size_t)nidx[pr] * (SH_DIM * NF);
    float ev0 = bf2f(sEmb[w][pp][lane]);
    float ev1 = bf2f(sEmb[w][pp][lane + 32]);
    for (int k = 0; k < SH_DIM; ++k) {
      int beg = sKoff[k], end = sKoff[k + 1];
      float acc0 = 0.0f, acc1 = 0.0f;
#pragma unroll 2
      for (int t = beg; t < end; ++t) {
        uint2 en = sEnt[t];
        int i = en.x & 31, j = (en.x >> 5) & 31;
        int di = (en.x >> 10) & 7, dj = (en.x >> 13) & 7;
        float cy = __uint_as_float(en.y) * sY[w][pp][i] * sY[w][pp][j];
        float av0 = bf2f(sA[w][di][pp][lane]);
        float bv0 = bf2f(sB[w][dj][pp][lane]);
        float av1 = bf2f(sA[w][di][pp][lane + 32]);
        float bv1 = bf2f(sB[w][dj][pp][lane + 32]);
        acc0 = fmaf(cy * av0, bv0, acc0);
        acc1 = fmaf(cy * av1, bv1, acc1);
      }
      atomicAdd(obase + k * NF + lane, ev0 * acc0);
      atomicAdd(obase + k * NF + lane + 32, ev1 * acc1);
    }
  }
}

// ======================= per-atom self term: out[:,0,:] += sE@Wt + bt =======================
__global__ __launch_bounds__(32 * NW, 1) void atom_kernel(
    const int* __restrict__ an, const float* __restrict__ species,
    const float* __restrict__ bt, const unsigned short* __restrict__ wtT,
    float* __restrict__ out, int num_atoms) {
  __shared__ unsigned short sTile[NW][16][RSTR];
  int w = threadIdx.x >> 5, lane = threadIdx.x & 31;
  int abase = (blockIdx.x * NW + w) * 16;
  int p2 = lane >> 1, half = lane & 1;
  int atom = min(abase + p2, num_atoms - 1);
  const float* se = species + an[atom] * NF + half * 32;
  {
    unsigned wb[16];
#pragma unroll
    for (int c = 0; c < 16; ++c) {
      unsigned lo = f2bf(se[2 * c]), hi = f2bf(se[2 * c + 1]);
      wb[c] = lo | (hi << 16);
    }
    uint4* dst = (uint4*)&sTile[w][p2][half * 32];
    dst[0] = make_uint4(wb[0], wb[1], wb[2], wb[3]);
    dst[1] = make_uint4(wb[4], wb[5], wb[6], wb[7]);
    dst[2] = make_uint4(wb[8], wb[9], wb[10], wb[11]);
    dst[3] = make_uint4(wb[12], wb[13], wb[14], wb[15]);
  }
  V16 a0, a1;
  load_afrag(&sTile[w][0][0], lane, 0, a0);
  load_afrag(&sTile[w][0][0], lane, 1, a1);
#pragma unroll
  for (int nt = 0; nt < 4; ++nt) {
    v8f c = {};
    V16 bf0, bf1;
    load_bfrag(wtT, lane, nt, 0, bf0);
    load_bfrag(wtT, lane, nt, 1, bf1);
    c = wmma_bf16(a0, bf0, c);
    c = wmma_bf16(a1, bf1, c);
    int n = nt * 16 + (lane & 15);
    float btv = bt[n];
#pragma unroll
    for (int v = 0; v < 8; ++v) {
      int at = abase + v + 8 * (lane >> 4);
      if (at < num_atoms) {
        float* o = out + (size_t)at * (SH_DIM * NF) + n;  // k==0 slab
        *o += c[v] + btv;
      }
    }
  }
}

// ======================= host driver =======================
extern "C" void kernel_launch(void* const* d_in, const int* in_sizes, int n_in,
                              void* d_out, int out_size, void* d_ws, size_t ws_size,
                              hipStream_t stream) {
  const int*   an      = (const int*)d_in[0];
  const float* disp    = (const float*)d_in[1];
  const int*   nidx    = (const int*)d_in[2];
  const float* species = (const float*)d_in[3];
  const float* Wt      = (const float*)d_in[4];
  const float* bt      = (const float*)d_in[5];
  const float* W1      = (const float*)d_in[6];
  const float* b1      = (const float*)d_in[7];
  const float* W2      = (const float*)d_in[8];
  const float* b2      = (const float*)d_in[9];
  int num_atoms = in_sizes[0];
  int num_pairs = in_sizes[1] / 3;
  float* out = (float*)d_out;
  (void)n_in; (void)ws_size;

  // workspace layout (all regions 16B aligned)
  char* ws = (char*)d_ws;
  int*   koff_g      = (int*)ws;                                   // 26 ints (pad 128B)
  uint2* ent_strided = (uint2*)(ws + 128);                         // 25*CAP_K*8 = 128000 B
  uint2* ent_packed  = (uint2*)(ws + 128 + 25 * CAP_K * 8);        // ENT_CAP*8 = 24576 B
  unsigned short* wt1T =
      (unsigned short*)(ws + 128 + 25 * CAP_K * 8 + ENT_CAP * 8);
  unsigned short* wt2T = wt1T + 5 * 4096;
  unsigned short* wtT  = wt2T + 5 * 4096;

  // 1) build CG sparse table on-device (uses d_out as transient scratch pre-memset)
  cg_dense_kernel<<<(SH_DIM * SH_DIM * SH_DIM + 255) / 256, 256, 0, stream>>>(out);
  cg_sparse_kernel<<<1, 32, 0, stream>>>(out, ent_strided, ent_packed, koff_g);
  // 2) bf16-transpose all weight matrices
  prep_weights_kernel<<<(5 * 4096 + 4096 + 255) / 256, 256, 0, stream>>>(
      W1, W2, Wt, wt1T, wt2T, wtT);
  // 3) zero output
  hipMemsetAsync(d_out, 0, (size_t)out_size * sizeof(float), stream);
  // 4) fused pair kernel (16 pairs per wave)
  int pblk = (num_pairs + 16 * NW - 1) / (16 * NW);
  pair_kernel<<<pblk, 32 * NW, 0, stream>>>(an, disp, nidx, species, bt, b1, b2,
                                            wt1T, wt2T, wtT, ent_packed, koff_g,
                                            out, num_pairs, num_atoms);
  // 5) per-atom self term into the l=0 slab
  int ablk = (num_atoms + 16 * NW - 1) / (16 * NW);
  atom_kernel<<<ablk, 32 * NW, 0, stream>>>(an, species, bt, wtT, out, num_atoms);
}
